// MultiHeadAttnRNN_27281632264821
// MI455X (gfx1250) — matrix operand order repaired and checked by reading, not compiled, not run on hardware
//
#include <hip/hip_runtime.h>
#include <hip/hip_bf16.h>
#include <cstdint>

// ---------------- problem constants (match reference) ----------------
#define BS     128
#define DMODEL 512
#define HDIM   512
#define VOUT   1000
#define VPAD   1024
#define MEMLEN 512
#define NHEADS 8
#define DK     64
#define TMAX   160
#define H3     1536
#define H2     1024

typedef __attribute__((ext_vector_type(16))) __bf16 v16bf;
typedef __attribute__((ext_vector_type(8)))  float  v8f;

// LDS byte address of a shared-memory pointer (generic addr low 32 bits == LDS offset)
__device__ __forceinline__ uint32_t lds_addr(const void* p) {
    return (uint32_t)(uintptr_t)p;
}

// CDNA5 async global->LDS copy, 16B per lane, tracked by ASYNCcnt.
__device__ __forceinline__ void async_load_b128(uint32_t lds_dst, uint32_t voff, const void* sbase) {
    asm volatile("global_load_async_to_lds_b128 %0, %1, %2 offset:0"
                 :
                 : "v"(lds_dst), "v"(voff), "s"(sbase)
                 : "memory");
}

// =====================================================================
// WMMA bf16 GEMM:  C[M,N] = A[M,K] * W[N,K]^T + bias
// 256 threads = 8 waves; block tile 128(M) x 64(N); K step 32.
// Double-buffered LDS tiles filled with GLOBAL_LOAD_ASYNC_TO_LDS_B128.
// =====================================================================
enum { EPI_F32 = 0, EPI_KV = 1, EPI_FC = 2 };

template<int MODE>
__global__ __launch_bounds__(256)
void gemm_wmma(const __bf16* __restrict__ A, int lda,   // M x K bf16 row-major
               const __bf16* __restrict__ W,            // N x K bf16 row-major
               const float*  __restrict__ bias, int nbias,
               float*  __restrict__ Cf, int ldc,        // EPI_F32 / EPI_FC out
               __bf16* __restrict__ Ckv,                // EPI_KV: [BS][H][MEM][DK]
               float*  __restrict__ outMasked,          // EPI_FC: d_out outputs slice
               const int* __restrict__ lens, int step,
               int M, int N, int K)
{
    const int ATS = 40;   // LDS row stride in bf16 (80B): 16B-aligned, bank-spread
    const int BTS = 40;
    __shared__ alignas(16) __bf16 At[2][128 * 40];   // 2 x 10240 B
    __shared__ alignas(16) __bf16 Bt[2][ 64 * 40];   // 2 x  5120 B

    const int tid  = threadIdx.x;
    const int wave = tid >> 5;
    const int lane = tid & 31;
    const int rowBase = blockIdx.y * 128;
    const int colBase = blockIdx.x * 64;

    // ---- async staging geometry ----
    // A tile 128x32 bf16 = 512 chunks of 16B: chunk e -> row e>>2, 16B-chunk e&3
    const int ar0 = tid >> 2,         ac0 = tid & 3;          // e = tid
    const int ar1 = (tid + 256) >> 2, ac1 = tid & 3;          // e = tid + 256
    // B tile 64x32 bf16 = 256 chunks: chunk tid -> row tid>>2, chunk tid&3
    const int bn = tid >> 2, bc = tid & 3;

    uint32_t ldsA0[2], ldsA1[2], ldsB[2];
    #pragma unroll
    for (int bb = 0; bb < 2; ++bb) {
        ldsA0[bb] = lds_addr(&At[bb][ar0 * ATS + ac0 * 8]);
        ldsA1[bb] = lds_addr(&At[bb][ar1 * ATS + ac1 * 8]);
        ldsB [bb] = lds_addr(&Bt[bb][bn  * BTS + bc  * 8]);
    }
    const __bf16* Abase = A + (size_t)rowBase * lda;
    const __bf16* Wbase = W + (size_t)colBase * K;
    const uint32_t voffA0 = (uint32_t)((ar0 * lda + ac0 * 8) * 2);
    const uint32_t voffA1 = (uint32_t)((ar1 * lda + ac1 * 8) * 2);
    const uint32_t voffB  = (uint32_t)((bn  * K   + bc  * 8) * 2);

    // prologue: stage K-slice 0 into buffer 0
    async_load_b128(ldsA0[0], voffA0, Abase);
    async_load_b128(ldsA1[0], voffA1, Abase);
    async_load_b128(ldsB [0], voffB,  Wbase);

    v8f acc[4];
    #pragma unroll
    for (int t = 0; t < 4; ++t)
        #pragma unroll
        for (int j = 0; j < 8; ++j) acc[t][j] = 0.0f;

    const int nk = K >> 5;
    for (int ks = 0; ks < nk; ++ks) {
        const int cb = ks & 1;
        if (ks + 1 < nk) {
            // issue next slice into the other buffer, then wait for current
            const uint32_t adv = (uint32_t)((ks + 1) * 64);   // 32 bf16 = 64B
            async_load_b128(ldsA0[cb ^ 1], voffA0 + adv, Abase);
            async_load_b128(ldsA1[cb ^ 1], voffA1 + adv, Abase);
            async_load_b128(ldsB [cb ^ 1], voffB  + adv, Wbase);
            asm volatile("s_wait_asynccnt 0x3" ::: "memory");  // current 3 done, next 3 in flight
        } else {
            asm volatile("s_wait_asynccnt 0x0" ::: "memory");
        }
        __syncthreads();

        // ---- A fragment (ISA 16-bit A 16x32 layout) ----
        const __bf16* arp = &At[cb][(wave * 16 + (lane & 15)) * ATS + (lane >> 4) * 8];
        v16bf a;
        *(uint4*)&a       = *(const uint4*)(arp);
        *((uint4*)&a + 1) = *(const uint4*)(arp + 16);

        // ---- preload all 4 B fragments, then back-to-back WMMAs ----
        v16bf bf[4];
        #pragma unroll
        for (int t = 0; t < 4; ++t) {
            const __bf16* brp = &Bt[cb][(t * 16 + (lane & 15)) * BTS + (lane >> 4) * 8];
            *(uint4*)&bf[t]       = *(const uint4*)(brp);
            *((uint4*)&bf[t] + 1) = *(const uint4*)(brp + 16);
        }
        #pragma unroll
        for (int t = 0; t < 4; ++t)
            acc[t] = __builtin_amdgcn_wmma_f32_16x16x32_bf16(
                         false, a, false, bf[t], (short)0, acc[t], false, false);
        __syncthreads();   // all waves done reading buf[cb] before its async overwrite
    }

    // ---- epilogue: D layout — VGPR i: lanes 0-15 -> M=i, lanes 16-31 -> M=8+i
    const int nlo = lane & 15;
    const int mhi = (lane >> 4) * 8;
    #pragma unroll
    for (int t = 0; t < 4; ++t) {
        int col = colBase + t * 16 + nlo;
        float bv = (col < nbias) ? bias[col] : 0.0f;
        #pragma unroll
        for (int i = 0; i < 8; ++i) {
            int row = rowBase + wave * 16 + mhi + i;
            float v = acc[t][i] + bv;
            if constexpr (MODE == EPI_F32) {
                Cf[(size_t)row * ldc + col] = v;
            } else if constexpr (MODE == EPI_KV) {
                // row = m*BS + b ; col = h*64 + d  ->  kp[b][h][m][d]
                int b  = row & (BS - 1);
                int mm = row >> 7;
                int h  = col >> 6;
                int d  = col & 63;
                Ckv[(((size_t)b * NHEADS + h) * MEMLEN + mm) * DK + d] = (__bf16)v;
            } else { // EPI_FC
                Cf[(size_t)row * ldc + col] = v;              // out_cur (unmasked carry)
                if (col < VOUT) {
                    float ov = (step < lens[row]) ? v : 0.0f;
                    outMasked[((size_t)step * BS + row) * VOUT + col] = ov;
                }
            }
        }
    }
}

// =====================================================================
// argmax (first-occurrence, matches jnp.argmax) + build xcat = [emb[tok], hid]
// =====================================================================
__global__ __launch_bounds__(256)
void argmax_concat(const float* __restrict__ out_cur,   // BS x VPAD
                   const __bf16* __restrict__ embb,     // V x H (bf16)
                   const __bf16* __restrict__ hidb,     // BS x H (bf16)
                   __bf16* __restrict__ xcat)           // BS x 2H (bf16)
{
    __shared__ float sv[256];
    __shared__ int   si[256];
    __shared__ int   stok;
    const int b = blockIdx.x, tid = threadIdx.x;

    float best = -INFINITY; int bidx = 0x7fffffff;
    for (int c = tid; c < VOUT; c += 256) {
        float v = out_cur[(size_t)b * VPAD + c];
        if (v > best) { best = v; bidx = c; }
    }
    sv[tid] = best; si[tid] = bidx;
    __syncthreads();
    for (int s = 128; s > 0; s >>= 1) {
        if (tid < s) {
            float v2 = sv[tid + s]; int i2 = si[tid + s];
            if (v2 > sv[tid] || (v2 == sv[tid] && i2 < si[tid])) { sv[tid] = v2; si[tid] = i2; }
        }
        __syncthreads();
    }
    if (tid == 0) stok = si[0];
    __syncthreads();
    const int tok = stok;
    for (int j = tid; j < HDIM; j += 256) {
        xcat[(size_t)b * H2 + j]        = embb[(size_t)tok * HDIM + j];
        xcat[(size_t)b * H2 + HDIM + j] = hidb[(size_t)b * HDIM + j];
    }
}

// =====================================================================
// attention for one (batch, head): scores -> softmax -> ctx
// kp/vp stay L2-resident (134MB bf16 < 192MB L2) — batched matvec on VALU.
// =====================================================================
__global__ __launch_bounds__(256)
void attention(const float*  __restrict__ q,     // BS x D (bias included)
               const __bf16* __restrict__ kp,    // [BS][H][MEM][DK]
               const __bf16* __restrict__ vp,
               float*  __restrict__ attw,        // [BS][H][MEM]
               __bf16* __restrict__ ctxb)        // BS x D
{
    __shared__ float qs[DK];
    __shared__ float sc[MEMLEN];
    __shared__ float red[256];
    const int bh = blockIdx.x;        // b*8 + h
    const int b = bh >> 3, h = bh & 7;
    const int tid = threadIdx.x;
    const float scale = 0.125f;       // 1/sqrt(64)

    if (tid < DK) qs[tid] = q[(size_t)b * DMODEL + h * DK + tid] * scale;
    __syncthreads();

    const __bf16* kbase = kp + (size_t)bh * MEMLEN * DK;
    float lmax = -INFINITY;
    for (int m = tid; m < MEMLEN; m += 256) {
        const __bf16* kr = kbase + (size_t)m * DK;
        float s = 0.0f;
        #pragma unroll 8
        for (int d = 0; d < DK; ++d) s += (float)kr[d] * qs[d];
        sc[m] = s;
        lmax = fmaxf(lmax, s);
    }
    red[tid] = lmax; __syncthreads();
    for (int s = 128; s > 0; s >>= 1) { if (tid < s) red[tid] = fmaxf(red[tid], red[tid + s]); __syncthreads(); }
    const float mx = red[0]; __syncthreads();

    float lsum = 0.0f;
    for (int m = tid; m < MEMLEN; m += 256) {
        float e = expf(sc[m] - mx);
        sc[m] = e; lsum += e;
    }
    red[tid] = lsum; __syncthreads();
    for (int s = 128; s > 0; s >>= 1) { if (tid < s) red[tid] += red[tid + s]; __syncthreads(); }
    const float inv = 1.0f / red[0]; __syncthreads();

    for (int m = tid; m < MEMLEN; m += 256) {
        float p = sc[m] * inv;
        sc[m] = p;
        attw[(size_t)bh * MEMLEN + m] = p;
    }
    __syncthreads();

    // ctx[d] = sum_m p[m] * vp[m][d]; 256 thr = 64 d x 4 m-chunks
    const __bf16* vbase = vp + (size_t)bh * MEMLEN * DK;
    const int d = tid & 63, chunk = tid >> 6;
    float partial = 0.0f;
    for (int m = chunk * 128; m < chunk * 128 + 128; ++m)
        partial += sc[m] * (float)vbase[(size_t)m * DK + d];
    red[tid] = partial; __syncthreads();
    if (tid < 64) {
        float tot = red[tid] + red[tid + 64] + red[tid + 128] + red[tid + 192];
        ctxb[(size_t)b * DMODEL + h * DK + d] = (__bf16)tot;
    }
}

// mean over heads -> masked scores output
__global__ void scores_mean(const float* __restrict__ attw, const int* __restrict__ lens,
                            int step, float* __restrict__ outS)
{
    int idx = blockIdx.x * blockDim.x + threadIdx.x;   // BS*MEM
    if (idx >= BS * MEMLEN) return;
    int b = idx >> 9, m = idx & 511;
    float s = 0.0f;
    #pragma unroll
    for (int h = 0; h < NHEADS; ++h) s += attw[((size_t)b * NHEADS + h) * MEMLEN + m];
    s *= (1.0f / NHEADS);
    outS[((size_t)step * BS + b) * MEMLEN + m] = (step < lens[b]) ? s : 0.0f;
}

// GRU gate fusion (PyTorch order r,z,n); writes carry (unmasked) + masked hiddens
__global__ void gru_gate(const float* __restrict__ gi, const float* __restrict__ gh,
                         float* __restrict__ hid, __bf16* __restrict__ hidb,
                         const int* __restrict__ lens, int step, float* __restrict__ outH)
{
    int idx = blockIdx.x * blockDim.x + threadIdx.x;   // BS*H
    if (idx >= BS * HDIM) return;
    int b = idx >> 9, j = idx & 511;
    size_t g = (size_t)b * H3;
    float ir = gi[g + j], iz = gi[g + HDIM + j], inn = gi[g + 2 * HDIM + j];
    float hr = gh[g + j], hz = gh[g + HDIM + j], hn  = gh[g + 2 * HDIM + j];
    float r = 1.0f / (1.0f + expf(-(ir + hr)));
    float z = 1.0f / (1.0f + expf(-(iz + hz)));
    float n = tanhf(inn + r * hn);
    float hnew = (1.0f - z) * n + z * hid[idx];
    hid[idx]  = hnew;
    hidb[idx] = (__bf16)hnew;
    outH[((size_t)step * BS + b) * HDIM + j] = (step < lens[b]) ? hnew : 0.0f;
}

__global__ void init_state(float* out_cur, float* hid, __bf16* hidb)
{
    int idx = blockIdx.x * blockDim.x + threadIdx.x;
    if (idx < BS * VPAD) out_cur[idx] = 0.0f;
    if (idx < BS * HDIM) { hid[idx] = 0.0f; hidb[idx] = (__bf16)0.0f; }
}

__global__ void f32_to_bf16(const float* __restrict__ src, __bf16* __restrict__ dst, int n)
{
    int i = blockIdx.x * blockDim.x + threadIdx.x;
    if (i < n) dst[i] = (__bf16)src[i];
}

// convert rows x cols f32 into padRows x cols bf16, zero-padding extra rows
__global__ void convert_pad_rows(const float* __restrict__ src, __bf16* __restrict__ dst,
                                 int rows, int cols, int padRows)
{
    int i = blockIdx.x * blockDim.x + threadIdx.x;
    if (i >= padRows * cols) return;
    int r = i / cols;
    dst[i] = (r < rows) ? (__bf16)src[i] : (__bf16)0.0f;
}

// =====================================================================
extern "C" void kernel_launch(void* const* d_in, const int* in_sizes, int n_in,
                              void* d_out, int out_size, void* d_ws, size_t ws_size,
                              hipStream_t stream)
{
    const float* memory = (const float*)d_in[0];
    const float* emb    = (const float*)d_in[1];
    const float* Wq     = (const float*)d_in[2];
    const float* bq     = (const float*)d_in[3];
    const float* Wk     = (const float*)d_in[4];
    const float* bk     = (const float*)d_in[5];
    const float* Wv     = (const float*)d_in[6];
    const float* bv     = (const float*)d_in[7];
    const float* W_ih   = (const float*)d_in[8];
    const float* W_hh   = (const float*)d_in[9];
    const float* b_ih   = (const float*)d_in[10];
    const float* b_hh   = (const float*)d_in[11];
    const float* Wfc    = (const float*)d_in[12];
    const float* bfc    = (const float*)d_in[13];
    const int*   lens   = (const int*)d_in[14];

    // d_out: outputs (T,BS,V) | hiddens (T,BS,H) | scores (T,BS,MEM)
    float* outO = (float*)d_out;
    float* outH = outO + (size_t)TMAX * BS * VOUT;
    float* outS = outH + (size_t)TMAX * BS * HDIM;

    // ---- workspace carve-out ----
    uintptr_t cur = (uintptr_t)d_ws;
    auto take = [&](size_t bytes) -> void* {
        cur = (cur + 255) & ~(uintptr_t)255;
        void* p = (void*)cur;
        cur += bytes;
        return p;
    };
    // persistent across whole launch
    __bf16* wq_b   = (__bf16*)take((size_t)DMODEL * H2 * 2);
    __bf16* wih_b  = (__bf16*)take((size_t)H3 * DMODEL * 2);
    __bf16* whh_b  = (__bf16*)take((size_t)H3 * HDIM * 2);
    __bf16* wfc_b  = (__bf16*)take((size_t)VPAD * HDIM * 2);
    __bf16* emb_b  = (__bf16*)take((size_t)VOUT * HDIM * 2);
    __bf16* wk_b   = (__bf16*)take((size_t)DMODEL * DMODEL * 2);
    __bf16* wv_b   = (__bf16*)take((size_t)DMODEL * DMODEL * 2);
    __bf16* kp     = (__bf16*)take((size_t)BS * NHEADS * MEMLEN * DK * 2);
    __bf16* vp     = (__bf16*)take((size_t)BS * NHEADS * MEMLEN * DK * 2);
    // decode-phase scratch; mem_b (setup-phase only) aliases this same region,
    // since KV projection completes (stream-ordered) before init_state runs.
    uintptr_t unionBase = cur;
    float*  out_cur= (float*)take((size_t)BS * VPAD * 4);
    float*  hid    = (float*)take((size_t)BS * HDIM * 4);
    __bf16* hid_b  = (__bf16*)take((size_t)BS * HDIM * 2);
    __bf16* xcat   = (__bf16*)take((size_t)BS * H2 * 2);
    float*  qbuf   = (float*)take((size_t)BS * DMODEL * 4);
    float*  gi     = (float*)take((size_t)BS * H3 * 4);
    float*  gh     = (float*)take((size_t)BS * H3 * 4);
    __bf16* ctx_b  = (__bf16*)take((size_t)BS * DMODEL * 2);
    float*  attw   = (float*)take((size_t)BS * NHEADS * MEMLEN * 4);
    __bf16* mem_b  = (__bf16*)(((unionBase + 255) & ~(uintptr_t)255)); // (MEM*BS) x D bf16

    auto cvt = [&](const float* s, __bf16* d, int n) {
        f32_to_bf16<<<(n + 255) / 256, 256, 0, stream>>>(s, d, n);
    };
    cvt(Wq,   wq_b,  DMODEL * H2);
    cvt(W_ih, wih_b, H3 * DMODEL);
    cvt(W_hh, whh_b, H3 * HDIM);
    cvt(emb,  emb_b, VOUT * HDIM);
    cvt(Wk,   wk_b,  DMODEL * DMODEL);
    cvt(Wv,   wv_b,  DMODEL * DMODEL);
    cvt(memory, mem_b, MEMLEN * BS * DMODEL);
    convert_pad_rows<<<(VPAD * HDIM + 255) / 256, 256, 0, stream>>>(Wfc, wfc_b, VOUT, HDIM, VPAD);

    // ---- one-time K/V projection: (MEM*BS) x 512 x 512, scatter to [b][h][m][d] ----
    {
        dim3 g(DMODEL / 64, (MEMLEN * BS) / 128);
        gemm_wmma<EPI_KV><<<g, 256, 0, stream>>>(
            mem_b, DMODEL, wk_b, bk, DMODEL, nullptr, 0, kp, nullptr, nullptr, 0,
            MEMLEN * BS, DMODEL, DMODEL);
        gemm_wmma<EPI_KV><<<g, 256, 0, stream>>>(
            mem_b, DMODEL, wv_b, bv, DMODEL, nullptr, 0, vp, nullptr, nullptr, 0,
            MEMLEN * BS, DMODEL, DMODEL);
    }

    // mem_b dead from here; decode scratch takes over the aliased region.
    init_state<<<(BS * VPAD + 255) / 256, 256, 0, stream>>>(out_cur, hid, hid_b);

    // ---- sequential decode ----
    for (int t = 0; t < TMAX; ++t) {
        argmax_concat<<<BS, 256, 0, stream>>>(out_cur, emb_b, hid_b, xcat);

        gemm_wmma<EPI_F32><<<dim3(DMODEL / 64, 1), 256, 0, stream>>>(
            xcat, H2, wq_b, bq, DMODEL, qbuf, DMODEL, nullptr, nullptr, nullptr, 0,
            BS, DMODEL, H2);

        attention<<<BS * NHEADS, 256, 0, stream>>>(qbuf, kp, vp, attw, ctx_b);
        scores_mean<<<(BS * MEMLEN) / 256, 256, 0, stream>>>(attw, lens, t, outS);

        gemm_wmma<EPI_F32><<<dim3(H3 / 64, 1), 256, 0, stream>>>(
            ctx_b, DMODEL, wih_b, b_ih, H3, gi, H3, nullptr, nullptr, nullptr, 0,
            BS, H3, DMODEL);
        gemm_wmma<EPI_F32><<<dim3(H3 / 64, 1), 256, 0, stream>>>(
            hid_b, HDIM, whh_b, b_hh, H3, gh, H3, nullptr, nullptr, nullptr, 0,
            BS, H3, HDIM);

        gru_gate<<<(BS * HDIM) / 256, 256, 0, stream>>>(gi, gh, hid, hid_b, lens, t, outH);

        gemm_wmma<EPI_FC><<<dim3(VPAD / 64, 1), 256, 0, stream>>>(
            hid_b, HDIM, wfc_b, bfc, VOUT, out_cur, VPAD, nullptr, outO, lens, t,
            BS, VPAD, HDIM);
    }
    (void)in_sizes; (void)n_in; (void)out_size; (void)ws_size;
}
